// TransformerBlock_76390288327336
// MI455X (gfx1250) — matrix operand order repaired and checked by reading, not compiled, not run on hardware
//
#include <hip/hip_runtime.h>
#include <hip/hip_bf16.h>
#include <math.h>

typedef _Float16 f16;
typedef __attribute__((ext_vector_type(16))) _Float16 v16h;
typedef __attribute__((ext_vector_type(8)))  float    v8f;

// ---------------- geometry ----------------
#define PL_   8
#define LAT_  96
#define LON_  180
#define C_    192
#define H_    6
#define D_    32
#define W0_   2
#define W1_   6
#define W2_   12
#define S0_   1
#define S1_   3
#define S2_   6
#define NPL_  4
#define NLAT_ 16
#define NLON_ 15
#define NW_   64
#define NT_   144          // tokens per window
#define LTOT_ 138240       // total tokens
#define NWIN_ 960          // NLON_*NW_
#define SCALE_ 0.17677669529663687f   // D^-0.5

// attention LDS strides (f16 units unless noted)
constexpr int QSTR  = 40;   // 144x32 q/k tiles (row-major, padded)
constexpr int VTSTR = 168;  // 32x160 v tile, TRANSPOSED [d][token], token-padded
constexpr int SSTR  = 168;  // 144x144 scores, f32
constexpr int PSTR  = 168;  // 144x160 probs, f16 (zero-padded cols 144..159)
constexpr int KP    = 160;
constexpr size_t ATTN_LDS =
    (size_t)NT_*QSTR*2 +      // sQ
    (size_t)NT_*QSTR*2 +      // sK
    (size_t)D_*VTSTR*2 +      // sVt (32 rows)
    (size_t)NT_*SSTR*4 +      // sS (f32)
    (size_t)NT_*PSTR*2 +      // sP
    (size_t)NT_*4 +           // region ids
    (size_t)NT_*4 +           // EA
    (size_t)NT_*4;            // EB

// ---------------- WMMA helpers ----------------
__device__ __forceinline__ v8f wmma_f16(v16h a, v16h b, v8f c) {
  // D = A(16x32 f16) * B(32x16 f16) + C(16x16 f32)
  return __builtin_amdgcn_wmma_f32_16x16x32_f16(false, a, false, b, (short)0, c, false, false);
}

// A fragment: 16x32 (MxK), row-major source, ld in f16 units (ld*2 must be mult of 16).
// Per-lane: two contiguous 8-f16 chunks at K = 8*half and K = 16+8*half, row m = lane&15.
__device__ __forceinline__ v16h load_frag_a(const f16* base, int ld, int lane) {
  int m = lane & 15, half = lane >> 4;
  const f16* row = base + (size_t)m * ld + 8*half;
  v16h a;
  *((uint4*)&a + 0) = *(const uint4*)(row + 0);
  *((uint4*)&a + 1) = *(const uint4*)(row + 16);
  return a;
}
// B fragment: 32x16 (KxN) from an N-MAJOR source (row n holds K contiguously).
// Per-lane: two contiguous 8-f16 chunks at K = 8*half and K = 16+8*half, row n = lane&15.
__device__ __forceinline__ v16h load_frag_ct(const f16* base, int ld, int lane) {
  const f16* row = base + (size_t)(lane & 15) * ld + 8 * (lane >> 4);
  v16h b;
  *((uint4*)&b + 0) = *(const uint4*)(row + 0);
  *((uint4*)&b + 1) = *(const uint4*)(row + 16);
  return b;
}

__device__ __forceinline__ float wave_sum(float v) {
#pragma unroll
  for (int off = 16; off > 0; off >>= 1) v += __shfl_xor(v, off, 32);
  return v;
}
__device__ __forceinline__ float wave_max(float v) {
#pragma unroll
  for (int off = 16; off > 0; off >>= 1) v = fmaxf(v, __shfl_xor(v, off, 32));
  return v;
}

// shift-window region id for token n of window (l, wi) -- replicates MASK construction
__device__ __forceinline__ int region_id(int l, int wi, int n) {
  int a2 = n % W2_, a1 = (n / W2_) % W1_, a0 = n / (W1_*W2_);
  int ipl = wi / NLAT_, ilat = wi % NLAT_;
  int pl  = ipl*W0_ + a0;     // rolled coords
  int lat = ilat*W1_ + a1;
  int lon = l*W2_ + a2;
  int r0 = (pl  < PL_  - W0_) ? 0 : (pl  < PL_  - S0_ ? 1 : 2);
  int r1 = (lat < LAT_ - W1_) ? 0 : (lat < LAT_ - S1_ ? 1 : 2);
  int r2 = (lon < LON_ - W2_) ? 0 : 1;   // third slice removed by [:LON]
  return r0*9 + r1*3 + r2;
}

__device__ __forceinline__ void ln_row(const float* __restrict__ row,
                                       const float* __restrict__ w,
                                       const float* __restrict__ b,
                                       f16* __restrict__ o, int lane) {
  float v[6]; float s = 0.f;
#pragma unroll
  for (int i = 0; i < 6; ++i) { v[i] = row[lane + 32*i]; s += v[i]; }
  s = wave_sum(s);
  float mu = s * (1.0f / C_);
  float qs = 0.f;
#pragma unroll
  for (int i = 0; i < 6; ++i) { float d = v[i]-mu; qs += d*d; }
  qs = wave_sum(qs);
  float inv = rsqrtf(qs * (1.0f / C_) + 1e-5f);
#pragma unroll
  for (int i = 0; i < 6; ++i) { int c = lane + 32*i; o[c] = (f16)(((v[i]-mu)*inv)*w[c] + b[c]); }
}

// ---------------- kernels ----------------

// convert weights f32 -> f16, TRANSPOSED to N-major [Nout][K] so B-fragments are K-contiguous
__global__ void k_cvt(const float* __restrict__ a, const float* __restrict__ b,
                      const float* __restrict__ c, const float* __restrict__ d,
                      f16* oa, f16* ob, f16* oc, f16* od) {
  int i = blockIdx.x * blockDim.x + threadIdx.x;
  int st = gridDim.x * blockDim.x;
  for (int t = i; t < 576*192; t += st) { int j = t / 192, k = t % 192; oa[t] = (f16)a[(size_t)k*576 + j]; }
  for (int t = i; t < 192*192; t += st) { int j = t / 192, k = t % 192; ob[t] = (f16)b[(size_t)k*192 + j]; }
  for (int t = i; t < 768*192; t += st) { int j = t / 192, k = t % 192; oc[t] = (f16)c[(size_t)k*768 + j]; }
  for (int t = i; t < 192*768; t += st) { int j = t / 768, k = t % 768; od[t] = (f16)d[(size_t)k*192 + j]; }
}

// LN1 + roll(-S) + window partition -> hwin f16 [LTOT_][C_], one wave per token
__global__ __launch_bounds__(256) void k_ln1_window(const float* __restrict__ x,
    const float* __restrict__ w, const float* __restrict__ b, f16* __restrict__ hwin) {
  int gw = (blockIdx.x * 256 + threadIdx.x) >> 5;
  int lane = threadIdx.x & 31;
  if (gw >= LTOT_) return;
  int win = gw / NT_, n = gw % NT_;
  int l = win / NW_, wi = win % NW_;
  int a2 = n % W2_, a1 = (n / W2_) % W1_, a0 = n / (W1_*W2_);
  int ipl = wi / NLAT_, ilat = wi % NLAT_;
  int pl  = (ipl*W0_  + a0 + S0_) % PL_;
  int lat = (ilat*W1_ + a1 + S1_) % LAT_;
  int lon = (l*W2_    + a2 + S2_) % LON_;
  const float* row = x + (size_t)((pl*LAT_ + lat)*LON_ + lon) * C_;
  ln_row(row, w, b, hwin + (size_t)gw * C_, lane);
}

// QKV GEMM: [LTOT_ x 192] @ [192 x 576] (B transposed N-major)
// block tile 128x64, whole 128x192 A panel staged in LDS once (no per-k barriers)
__global__ __launch_bounds__(256) void k_qkv(const f16* __restrict__ A, const f16* __restrict__ Bt,
    const float* __restrict__ bias, f16* __restrict__ q, f16* __restrict__ k, f16* __restrict__ v) {
  const int ASTR = 200;                 // 192 + 8 pad, 400B rows (16B aligned)
  __shared__ f16 sA[128*200];
  __shared__ int sRow[128];             // win*H*NT*D + n*D per local row
  int mblk = blockIdx.x, nblk = blockIdx.y;
  int tid = threadIdx.x, lane = tid & 31, wv = tid >> 5;
  int m0b = mblk * 128;
  for (int idx = tid; idx < 128*12; idx += 256) {
    int r = idx / 12, cc = (idx % 12) * 16;
    *(uint4*)&sA[r*ASTR + cc] = *(const uint4*)(A + (size_t)(m0b + r) * C_ + cc);
  }
  if (tid < 128) {
    int m = m0b + tid;
    int win = m / NT_, n = m % NT_;
    sRow[tid] = win * (H_*NT_*D_) + n * D_;
  }
  __syncthreads();
  v8f acc[4] = {};
  for (int ks = 0; ks < 6; ++ks) {
    v16h a = load_frag_a(&sA[(wv*16)*ASTR + ks*32], ASTR, lane);
#pragma unroll
    for (int sub = 0; sub < 4; ++sub) {
      v16h b = load_frag_ct(Bt + (size_t)(nblk*64 + sub*16) * C_ + ks*32, C_, lane);
      acc[sub] = wmma_f16(a, b, acc[sub]);
    }
  }
  int half = lane >> 4, col = lane & 15;
#pragma unroll
  for (int sub = 0; sub < 4; ++sub) {
    int j = nblk*64 + sub*16 + col;
    int t = j / C_, rem = j % C_, hh = rem / D_, dd = rem % D_;
    float bj = bias[j];
    f16* dst = (t == 0) ? q : (t == 1) ? k : v;
    int hoff = hh * (NT_*D_) + dd;
#pragma unroll
    for (int r2 = 0; r2 < 8; ++r2) {
      int lm = wv*16 + r2 + 8*half;
      float val = acc[sub][r2] + bj;
      if (t == 0) val *= SCALE_;
      dst[(size_t)sRow[lm] + hoff] = (f16)val;
    }
  }
}

// per-window attention: S=QK^T + bias + shift-mask, softmax, O=PV. One workgroup per window.
__global__ __launch_bounds__(256) void k_attn(const f16* __restrict__ q, const f16* __restrict__ k,
    const f16* __restrict__ v, const float* __restrict__ bt, f16* __restrict__ o) {
  extern __shared__ char smem[];
  f16*   sQ   = (f16*)smem;
  f16*   sK   = sQ + NT_*QSTR;
  f16*   sVt  = sK + NT_*QSTR;            // transposed: [d][token], token padded to 160
  float* sS   = (float*)(sVt + D_*VTSTR);
  f16*   sP   = (f16*)(sS + NT_*SSTR);
  int*   sreg = (int*)(sP + NT_*PSTR);
  int*   sEA  = sreg + NT_;               // epi(n,m) = sEA[n] + sEB[m]
  int*   sEB  = sEA + NT_;

  int win = blockIdx.x, l = win / NW_, wi = win % NW_;
  int tid = threadIdx.x, lane = tid & 31, wv = tid >> 5;

  for (int n = tid; n < NT_; n += 256) {
    sreg[n] = region_id(l, wi, n);
    int a2 = n % W2_, a1 = (n / W2_) % W1_, a0 = n / (W1_*W2_);
    sEA[n] = 828*a0 + 23*a1 + a2 + (W2_-1);
    sEB[n] = 1656*a0 + 138*a1 + 12*a2;
  }
  const float* btw = bt + (size_t)wi * H_;   // + epi*(NW_*H_) + h

  for (int h = 0; h < H_; ++h) {
    __syncthreads();
    const f16* qh = q + (size_t)(win*H_ + h) * NT_ * D_;
    const f16* kh = k + (size_t)(win*H_ + h) * NT_ * D_;
    const f16* vh = v + (size_t)(win*H_ + h) * NT_ * D_;
    // Q/K panels: CDNA5 async global->LDS copies (ASYNCcnt-tracked, no VGPR round-trip).
    // Dynamic LDS starts at offset 0 (no static __shared__ in this kernel), so LDS byte
    // offsets are computed directly from the manual smem layout.
    for (int idx = tid; idx < NT_*2; idx += 256) {
      int r = idx >> 1, c = (idx & 1) * 16;
      unsigned lq = (unsigned)((r*QSTR + c) * 2);
      unsigned lk = (unsigned)((NT_*QSTR + r*QSTR + c) * 2);
      const f16* gq = &qh[r*D_ + c];
      const f16* gk = &kh[r*D_ + c];
      asm volatile("global_load_async_to_lds_b128 %0, %1, off"
                   :: "v"(lq), "v"(gq) : "memory");
      asm volatile("global_load_async_to_lds_b128 %0, %1, off"
                   :: "v"(lk), "v"(gk) : "memory");
    }
    // V: load rows vectorized, scatter transposed into sVt[d][token]
    for (int idx = tid; idx < NT_*4; idx += 256) {
      int r = idx >> 2, c = (idx & 3) * 8;
      uint4 pk = *(const uint4*)&vh[r*D_ + c];
      const f16* pf = (const f16*)&pk;
#pragma unroll
      for (int i = 0; i < 8; ++i) sVt[(c + i)*VTSTR + r] = pf[i];
    }
    for (int idx = tid; idx < D_*16; idx += 256) {      // zero K-padding tokens 144..159
      int dd = idx >> 4, t = NT_ + (idx & 15);
      sVt[dd*VTSTR + t] = (f16)0.f;
    }
    asm volatile("s_wait_asynccnt 0x0" ::: "memory");   // async copies landed in LDS
    __syncthreads();
    // S = Q K^T : 9x9 tiles, K=D=32 -> one WMMA per tile
    for (int tile = wv; tile < 81; tile += 8) {
      int ti = tile / 9, tj = tile % 9;
      v16h a = load_frag_a(&sQ[ti*16*QSTR], QSTR, lane);
      v16h b = load_frag_ct(&sK[tj*16*QSTR], QSTR, lane);
      v8f acc = {};
      acc = wmma_f16(a, b, acc);
      int half = lane >> 4, mk = tj*16 + (lane & 15);
      int eb = sEB[mk], rg = sreg[mk];
#pragma unroll
      for (int r = 0; r < 8; ++r) {
        int nq = ti*16 + r + 8*half;
        float sv = acc[r] + btw[(size_t)(sEA[nq] + eb) * (NW_*H_) + h];
        if (sreg[nq] != rg) sv -= 100.0f;
        sS[nq*SSTR + mk] = sv;
      }
    }
    __syncthreads();
    // softmax (fp32): one wave per row, lanes cooperate via shuffles
    for (int row = wv; row < NT_; row += 8) {
      float* rs = &sS[row*SSTR];
      float mx = -1e30f;
      for (int m = lane; m < NT_; m += 32) mx = fmaxf(mx, rs[m]);
      mx = wave_max(mx);
      float sum = 0.f;
      for (int m = lane; m < NT_; m += 32) { float e = __expf(rs[m] - mx); rs[m] = e; sum += e; }
      sum = wave_sum(sum);
      float rinv = 1.0f / sum;
      f16* pr = &sP[row*PSTR];
      for (int m = lane; m < NT_; m += 32) pr[m] = (f16)(rs[m] * rinv);
      if (lane < KP - NT_) pr[NT_ + lane] = (f16)0.f;    // zero K-padding
    }
    __syncthreads();
    // O = P V : 9x2 tiles, K padded 144->160 (5 k-steps); V transposed -> contiguous B frags
    for (int tile = wv; tile < 18; tile += 8) {
      int ti = tile >> 1, tj = tile & 1;
      v8f acc = {};
#pragma unroll
      for (int ks = 0; ks < 5; ++ks) {
        v16h a = load_frag_a(&sP[ti*16*PSTR + ks*32], PSTR, lane);
        v16h b = load_frag_ct(&sVt[(size_t)(tj*16)*VTSTR + ks*32], VTSTR, lane);
        acc = wmma_f16(a, b, acc);
      }
      int half = lane >> 4, d = tj*16 + (lane & 15);
#pragma unroll
      for (int r = 0; r < 8; ++r) {
        int n = ti*16 + r + 8*half;
        o[((size_t)win*NT_ + n)*C_ + h*D_ + d] = (f16)acc[r];
      }
    }
  }
}

// proj GEMM (B transposed) + un-window + un-roll + residual: xnew = x + scatter(o @ Wp + bp)
// block tile 128x64, whole A panel staged once, destination rows precomputed
__global__ __launch_bounds__(256) void k_proj(const f16* __restrict__ A, const f16* __restrict__ Bt,
    const float* __restrict__ bias, const float* __restrict__ xin, float* __restrict__ xnew) {
  const int ASTR = 200;
  __shared__ f16 sA[128*200];
  __shared__ int sDst[128];
  int mblk = blockIdx.x, nblk = blockIdx.y;
  int tid = threadIdx.x, lane = tid & 31, wv = tid >> 5;
  int m0b = mblk * 128;
  for (int idx = tid; idx < 128*12; idx += 256) {
    int r = idx / 12, cc = (idx % 12) * 16;
    *(uint4*)&sA[r*ASTR + cc] = *(const uint4*)(A + (size_t)(m0b + r) * C_ + cc);
  }
  if (tid < 128) {
    int m = m0b + tid;
    int win = m / NT_, n = m % NT_;
    int l = win / NW_, wi = win % NW_;
    int a2 = n % W2_, a1 = (n / W2_) % W1_, a0 = n / (W1_*W2_);
    int ipl = wi / NLAT_, ilat = wi % NLAT_;
    int pl  = (ipl*W0_  + a0 + S0_) % PL_;
    int lat = (ilat*W1_ + a1 + S1_) % LAT_;
    int lon = (l*W2_    + a2 + S2_) % LON_;
    sDst[tid] = (pl*LAT_ + lat)*LON_ + lon;
  }
  __syncthreads();
  v8f acc[4] = {};
  for (int ks = 0; ks < 6; ++ks) {
    v16h a = load_frag_a(&sA[(wv*16)*ASTR + ks*32], ASTR, lane);
#pragma unroll
    for (int sub = 0; sub < 4; ++sub) {
      v16h b = load_frag_ct(Bt + (size_t)(nblk*64 + sub*16) * C_ + ks*32, C_, lane);
      acc[sub] = wmma_f16(a, b, acc[sub]);
    }
  }
  int half = lane >> 4, col = lane & 15;
#pragma unroll
  for (int sub = 0; sub < 4; ++sub) {
    int j = nblk*64 + sub*16 + col;
    float bj = bias[j];
#pragma unroll
    for (int r2 = 0; r2 < 8; ++r2) {
      int lm = wv*16 + r2 + 8*half;
      size_t dr = (size_t)sDst[lm];
      xnew[dr*C_ + j] = acc[sub][r2] + bj + xin[dr*C_ + j];
    }
  }
}

// LN2 over xnew -> h2 f16
__global__ __launch_bounds__(256) void k_ln2(const float* __restrict__ x,
    const float* __restrict__ w, const float* __restrict__ b, f16* __restrict__ out) {
  int gw = (blockIdx.x * 256 + threadIdx.x) >> 5;
  int lane = threadIdx.x & 31;
  if (gw >= LTOT_) return;
  ln_row(x + (size_t)gw * C_, w, b, out + (size_t)gw * C_, lane);
}

// fused FFN: per 16-row tile: (h2 @ W1 + b1) -> exact GELU in LDS -> @ W2 + b2 + residual -> out
// both weight matrices transposed N-major
__global__ __launch_bounds__(256) void k_ffn(const f16* __restrict__ h2, const f16* __restrict__ w1t,
    const float* __restrict__ b1, const f16* __restrict__ w2t, const float* __restrict__ b2,
    const float* __restrict__ xres, float* __restrict__ out) {
  const int A1S = 200, G1S = 776;
  __shared__ f16 sA[16*200];
  __shared__ f16 sG[16*776];
  int m0 = blockIdx.x * 16;
  int tid = threadIdx.x, lane = tid & 31, wv = tid >> 5;
  if (tid < 192) {
    int r = tid / 12, cc = (tid % 12) * 16;
    *(uint4*)&sA[r*A1S + cc] = *(const uint4*)(h2 + (size_t)(m0 + r) * C_ + cc);
  }
  __syncthreads();
  // GEMM1: 16x768, K=192
  for (int nt = wv; nt < 48; nt += 8) {
    v8f acc = {};
#pragma unroll
    for (int ks = 0; ks < 6; ++ks) {
      v16h a = load_frag_a(&sA[ks*32], A1S, lane);
      v16h b = load_frag_ct(w1t + (size_t)(nt*16) * C_ + ks*32, C_, lane);
      acc = wmma_f16(a, b, acc);
    }
    int half = lane >> 4, col = lane & 15, j = nt*16 + col;
    float bj = b1[j];
#pragma unroll
    for (int r = 0; r < 8; ++r) {
      float vv = acc[r] + bj;
      float g = 0.5f * vv * (1.0f + erff(vv * 0.70710678118654752f));  // exact GELU
      sG[(r + 8*half)*G1S + j] = (f16)g;
    }
  }
  __syncthreads();
  // GEMM2: 16x192, K=768
  for (int nt = wv; nt < 12; nt += 8) {
    v8f acc = {};
#pragma unroll
    for (int ks = 0; ks < 24; ++ks) {
      v16h a = load_frag_a(&sG[ks*32], G1S, lane);
      v16h b = load_frag_ct(w2t + (size_t)(nt*16) * 768 + ks*32, 768, lane);
      acc = wmma_f16(a, b, acc);
    }
    int half = lane >> 4, col = lane & 15, j = nt*16 + col;
    float bj = b2[j];
#pragma unroll
    for (int r = 0; r < 8; ++r) {
      size_t row = (size_t)(m0 + r + 8*half);
      out[row*C_ + j] = acc[r] + bj + xres[row*C_ + j];
    }
  }
}

// ---------------- launcher ----------------
extern "C" void kernel_launch(void* const* d_in, const int* in_sizes, int n_in,
                              void* d_out, int out_size, void* d_ws, size_t ws_size,
                              hipStream_t stream) {
  const float* x      = (const float*)d_in[0];
  const float* ln1_w  = (const float*)d_in[1];
  const float* ln1_b  = (const float*)d_in[2];
  const float* qkv_w  = (const float*)d_in[3];
  const float* qkv_b  = (const float*)d_in[4];
  const float* proj_w = (const float*)d_in[5];
  const float* proj_b = (const float*)d_in[6];
  const float* bias_t = (const float*)d_in[7];
  const float* ln2_w  = (const float*)d_in[8];
  const float* ln2_b  = (const float*)d_in[9];
  const float* fc1_w  = (const float*)d_in[10];
  const float* fc1_b  = (const float*)d_in[11];
  const float* fc2_w  = (const float*)d_in[12];
  const float* fc2_b  = (const float*)d_in[13];
  float* out = (float*)d_out;
  (void)in_sizes; (void)n_in; (void)out_size; (void)ws_size;

  char* ws = (char*)d_ws;
  size_t off = 0;
  auto take = [&](size_t bytes) -> char* {
    char* p = ws + off;
    off = (off + bytes + 255) & ~(size_t)255;
    return p;
  };
  f16* qkvw16 = (f16*)take((size_t)576*192*2);   // transposed [576][192]
  f16* projw16 = (f16*)take((size_t)192*192*2);  // transposed [192][192]
  f16* fc1w16 = (f16*)take((size_t)768*192*2);   // transposed [768][192]
  f16* fc2w16 = (f16*)take((size_t)192*768*2);   // transposed [192][768]
  const size_t HS = (size_t)LTOT_ * C_ * 2;      // 53 MB, f16 LxC
  f16* hwin = (f16*)take(HS);                    // LN1+windowed input; reused as attention output
  f16* qbuf = (f16*)take(HS);
  f16* kbuf = (f16*)take(HS);
  f16* vbuf = (f16*)take(HS);
  f16*   owin = hwin;           // hwin dead after QKV
  float* xnew = (float*)qbuf;   // q+k dead after attention; spans qbuf..kbuf (2*HS = L*C*f32)
  f16*   h2   = vbuf;           // v dead after attention

  k_cvt<<<432, 256, 0, stream>>>(qkv_w, proj_w, fc1_w, fc2_w, qkvw16, projw16, fc1w16, fc2w16);
  k_ln1_window<<<LTOT_/8, 256, 0, stream>>>(x, ln1_w, ln1_b, hwin);
  k_qkv<<<dim3(LTOT_/128, 9), 256, 0, stream>>>(hwin, qkvw16, qkv_b, qbuf, kbuf, vbuf);
  k_attn<<<NWIN_, 256, (unsigned)ATTN_LDS, stream>>>(qbuf, kbuf, vbuf, bias_t, owin);
  k_proj<<<dim3(LTOT_/128, 3), 256, 0, stream>>>(owin, projw16, proj_b, x, xnew);
  k_ln2<<<LTOT_/8, 256, 0, stream>>>(xnew, ln2_w, ln2_b, h2);
  k_ffn<<<LTOT_/16, 256, 0, stream>>>(h2, fc1w16, fc1_b, fc2w16, fc2_b, xnew, out);
}
